// CDCOR_51625506897950
// MI455X (gfx1250) — compile-verified, matrix operand order checked by reading, not run on hardware
//
#include <hip/hip_runtime.h>

typedef __attribute__((ext_vector_type(16))) _Float16 v16h;
typedef __attribute__((ext_vector_type(8)))  _Float16 v8h;
typedef __attribute__((ext_vector_type(8)))  float    v8f;

#define DD 256
#define TM 16

__device__ __forceinline__ v16h cat16(v8h lo, v8h hi) {
  return __builtin_shufflevector(lo, hi, 0,1,2,3,4,5,6,7,8,9,10,11,12,13,14,15);
}

// One 16x16 output tile, accumulate over KD (multiple of 32).
// aPtr: &A_lds[lane&15][koff], row stride = KD-contiguous (K-major rows)
// bPtr: &W_T[n][koff], W_T is [N][K] row-major (so K contiguous per n)
template<int KD>
__device__ __forceinline__ v8f mma_chain(const _Float16* aPtr, const _Float16* bPtr, v8f acc) {
  #pragma unroll
  for (int k0 = 0; k0 < KD; k0 += 32) {
    v8h alo = *(const v8h*)(aPtr + k0);
    v8h ahi = *(const v8h*)(aPtr + k0 + 16);
    v8h blo = *(const v8h*)(bPtr + k0);
    v8h bhi = *(const v8h*)(bPtr + k0 + 16);
    acc = __builtin_amdgcn_wmma_f32_16x16x32_f16(
        false, cat16(alo, ahi), false, cat16(blo, bhi),
        (short)0, acc, false, false);
  }
  return acc;
}

__global__ void adj_prep_kernel(const float* __restrict__ adj,
                                float* __restrict__ adj_t_out,
                                _Float16* __restrict__ adjT_h) {
  int idx = blockIdx.x * blockDim.x + threadIdx.x;
  if (idx >= 512 * 512) return;
  int k = idx >> 9, n = idx & 511;
  float v = adj[idx];
  float vt = (fabsf(v) < 1e-4f) ? 0.0f : v;
  adj_t_out[idx] = vt;
  adjT_h[n * 512 + k] = (_Float16)vt;
}

__global__ void cvt_transpose_kernel(const float* __restrict__ src,
                                     _Float16* __restrict__ dst, int K, int N) {
  int idx = blockIdx.x * blockDim.x + threadIdx.x;
  if (idx >= K * N) return;
  int k = idx / N, n = idx % N;
  dst[n * K + k] = (_Float16)src[idx];
}

__global__ void __launch_bounds__(256)
cdcor_branch_kernel(const int* __restrict__ u_idx, const int* __restrict__ i_idx,
                    const float* __restrict__ user_table, const float* __restrict__ item_table,
                    const _Float16* __restrict__ eWT, const float* __restrict__ e_b,
                    const _Float16* __restrict__ ecWT, const float* __restrict__ ec_b,
                    const _Float16* __restrict__ adjT,
                    const _Float16* __restrict__ trWT, const float* __restrict__ tr_b,
                    const float* __restrict__ pr_W, const float* __restrict__ pr_b,
                    const _Float16* __restrict__ cl1WT, const float* __restrict__ cl1_b,
                    const float* __restrict__ cl2_W, const float* __restrict__ cl2_b,
                    float* __restrict__ out_pred, float* __restrict__ out_cls,
                    float* __restrict__ out_inp, float* __restrict__ out_causal) {
  __shared__ __align__(16) _Float16 attrH[TM][DD];
  __shared__ __align__(16) _Float16 cembH[TM][DD];
  __shared__ __align__(16) _Float16 embH[TM][DD];
  __shared__ __align__(16) _Float16 prefH[TM][DD];
  __shared__ __align__(16) float itemF[TM][DD];
  __shared__ float predAcc[TM][2];
  __shared__ float clsAcc[TM][2];

  const int tid  = threadIdx.x;
  const int lane = tid & 31;
  const int wv   = tid >> 5;
  const int n15  = lane & 15;
  const int mhi  = (lane >> 4) << 3;  // 0 or 8 : C/D row offset
  const int koff = (lane >> 4) << 3;  // 0 or 8 : A/B K offset for this lane half
  const int row0 = blockIdx.x * TM;

  // ---- stage 1: gather attr/item, convert attr->f16, write inp[:,0:256], init acc ----
  {
    const int r   = tid >> 4;
    const int seg = (tid & 15) << 4;
    const int u  = u_idx[row0 + r];
    const int it = i_idx[row0 + r];
    const float* us = user_table + (size_t)u * DD + seg;
    const float* is = item_table + (size_t)it * DD + seg;
    float* oi = out_inp + (size_t)(row0 + r) * (2 * DD) + seg;
    #pragma unroll
    for (int j = 0; j < 16; ++j) {
      float v = us[j];
      oi[j] = v;
      attrH[r][seg + j] = (_Float16)v;
      itemF[r][seg + j] = is[j];
    }
    if (tid < 32) {
      int m = tid >> 1, jj = tid & 1;
      predAcc[m][jj] = pr_b[jj];
      clsAcc[m][jj]  = cl2_b[jj];
    }
  }
  __syncthreads();

  const _Float16* aAttr = &attrH[n15][koff];

  // ---- stage 2: emb & c_emb (32 tiles, 4 per wave) ----
  for (int i = 0; i < 4; ++i) {
    int t = wv * 4 + i;          // uniform per wave
    bool isC = (t < 16);
    int nt = isC ? t : (t - 16);
    int n = nt * 16 + n15;
    const _Float16* bPtr = (isC ? ecWT : eWT) + (size_t)n * DD + koff;
    v8f acc = {};
    acc = mma_chain<DD>(aAttr, bPtr, acc);
    float bias = isC ? ec_b[n] : e_b[n];
    #pragma unroll
    for (int r = 0; r < 8; ++r) {
      int m = r + mhi;
      float v = acc[r] + bias;
      v = v > 0.0f ? v : 0.0f;  // relu
      if (isC) {
        out_inp[(size_t)(row0 + m) * (2 * DD) + DD + n] = v;
        cembH[m][n] = (_Float16)v;
      } else {
        embH[m][n] = (_Float16)v;
      }
    }
  }
  __syncthreads();

  const _Float16* aCemb = &cembH[n15][koff];
  const _Float16* aEmb  = &embH[n15][koff];

  // ---- stage 3: causal = [attr|c_emb] @ adj_t (32 tiles over N=512) ----
  for (int i = 0; i < 4; ++i) {
    int nt = wv * 4 + i;
    int n = nt * 16 + n15;       // 0..511
    const _Float16* bPtr = adjT + (size_t)n * (2 * DD) + koff;
    v8f acc = {};
    acc = mma_chain<DD>(aAttr, bPtr, acc);
    acc = mma_chain<DD>(aCemb, bPtr + DD, acc);
    #pragma unroll
    for (int r = 0; r < 8; ++r) {
      int m = r + mhi;
      float v = acc[r];
      out_causal[(size_t)(row0 + m) * (2 * DD) + n] = v;
      if (n >= DD) prefH[m][n - DD] = (_Float16)v;  // pref = causal[:, 256:512]
    }
  }
  __syncthreads();

  const _Float16* aPref = &prefH[n15][koff];

  // ---- stage 4: user = [emb|pref]@tr_W + tr_b ; pred = (user*item)@pr_W + pr_b ----
  for (int i = 0; i < 2; ++i) {
    int nt = wv * 2 + i;
    int n = nt * 16 + n15;       // 0..255
    const _Float16* bPtr = trWT + (size_t)n * (2 * DD) + koff;
    v8f acc = {};
    acc = mma_chain<DD>(aEmb, bPtr, acc);
    acc = mma_chain<DD>(aPref, bPtr + DD, acc);
    float tb = tr_b[n];
    float w0 = pr_W[n * 2 + 0];
    float w1 = pr_W[n * 2 + 1];
    #pragma unroll
    for (int r = 0; r < 8; ++r) {
      int m = r + mhi;
      float ui = (acc[r] + tb) * itemF[m][n];
      float p0 = ui * w0, p1 = ui * w1;
      #pragma unroll
      for (int off = 8; off > 0; off >>= 1) {
        p0 += __shfl_xor(p0, off);
        p1 += __shfl_xor(p1, off);
      }
      if (n15 == 0) {
        atomicAdd(&predAcc[m][0], p0);
        atomicAdd(&predAcc[m][1], p1);
      }
    }
  }

  // ---- stage 5: cls = sigmoid(c_emb@cl1_W + cl1_b) @ cl2_W + cl2_b (8 tiles, N=128) ----
  {
    int n = wv * 16 + n15;       // 0..127
    const _Float16* bPtr = cl1WT + (size_t)n * DD + koff;
    v8f acc = {};
    acc = mma_chain<DD>(aCemb, bPtr, acc);
    float hb = cl1_b[n];
    float w0 = cl2_W[n * 2 + 0];
    float w1 = cl2_W[n * 2 + 1];
    #pragma unroll
    for (int r = 0; r < 8; ++r) {
      int m = r + mhi;
      float h = 1.0f / (1.0f + __expf(-(acc[r] + hb)));
      float c0 = h * w0, c1 = h * w1;
      #pragma unroll
      for (int off = 8; off > 0; off >>= 1) {
        c0 += __shfl_xor(c0, off);
        c1 += __shfl_xor(c1, off);
      }
      if (n15 == 0) {
        atomicAdd(&clsAcc[m][0], c0);
        atomicAdd(&clsAcc[m][1], c1);
      }
    }
  }
  __syncthreads();

  if (tid < 32) {
    int m = tid >> 1, jj = tid & 1;
    out_pred[(size_t)(row0 + m) * 2 + jj] = predAcc[m][jj];
    out_cls[(size_t)(row0 + m) * 2 + jj]  = clsAcc[m][jj];
  }
}

extern "C" void kernel_launch(void* const* d_in, const int* in_sizes, int n_in,
                              void* d_out, int out_size, void* d_ws, size_t ws_size,
                              hipStream_t stream) {
  const int*   s_u = (const int*)d_in[0];
  const int*   t_u = (const int*)d_in[1];
  const int*   s_i = (const int*)d_in[2];
  const int*   t_i = (const int*)d_in[3];
  const float* s_user_table = (const float*)d_in[4];
  const float* t_user_table = (const float*)d_in[5];
  const float* s_item_table = (const float*)d_in[6];
  const float* t_item_table = (const float*)d_in[7];
  const float* es_W  = (const float*)d_in[8];
  const float* es_b  = (const float*)d_in[9];
  const float* et_W  = (const float*)d_in[10];
  const float* et_b  = (const float*)d_in[11];
  const float* ec_W  = (const float*)d_in[12];
  const float* ec_b  = (const float*)d_in[13];
  const float* adj   = (const float*)d_in[14];
  const float* cl1_W = (const float*)d_in[15];
  const float* cl1_b = (const float*)d_in[16];
  const float* cl2_W = (const float*)d_in[17];
  const float* cl2_b = (const float*)d_in[18];
  const float* st_W  = (const float*)d_in[19];
  const float* st_b  = (const float*)d_in[20];
  const float* tt_W  = (const float*)d_in[21];
  const float* tt_b  = (const float*)d_in[22];
  const float* sp_W  = (const float*)d_in[23];
  const float* sp_b  = (const float*)d_in[24];
  const float* tp_W  = (const float*)d_in[25];
  const float* tp_b  = (const float*)d_in[26];

  // d_out layout: s_pred, t_pred, s_cls, t_cls, s_inp, t_inp, s_causal, t_causal, adj_t
  float* out = (float*)d_out;
  float* o_s_pred   = out;
  float* o_t_pred   = o_s_pred + 131072;
  float* o_s_cls    = o_t_pred + 131072;
  float* o_t_cls    = o_s_cls  + 131072;
  float* o_s_inp    = o_t_cls  + 131072;
  float* o_t_inp    = o_s_inp    + 33554432;
  float* o_s_causal = o_t_inp    + 33554432;
  float* o_t_causal = o_s_causal + 33554432;
  float* o_adj      = o_t_causal + 33554432;

  // workspace: transposed f16 weights, W^T[N][K]
  _Float16* ws    = (_Float16*)d_ws;
  _Float16* adjT  = ws;               // 512*512
  _Float16* esWT  = adjT + 262144;    // 256*256
  _Float16* etWT  = esWT + 65536;
  _Float16* ecWT  = etWT + 65536;
  _Float16* stWT  = ecWT + 65536;     // 256*512
  _Float16* ttWT  = stWT + 131072;
  _Float16* cl1WT = ttWT + 131072;    // 128*256

  adj_prep_kernel<<<1024, 256, 0, stream>>>(adj, o_adj, adjT);
  cvt_transpose_kernel<<<256, 256, 0, stream>>>(es_W, esWT, 256, 256);
  cvt_transpose_kernel<<<256, 256, 0, stream>>>(et_W, etWT, 256, 256);
  cvt_transpose_kernel<<<256, 256, 0, stream>>>(ec_W, ecWT, 256, 256);
  cvt_transpose_kernel<<<512, 256, 0, stream>>>(st_W, stWT, 512, 256);
  cvt_transpose_kernel<<<512, 256, 0, stream>>>(tt_W, ttWT, 512, 256);
  cvt_transpose_kernel<<<128, 256, 0, stream>>>(cl1_W, cl1WT, 256, 128);

  cdcor_branch_kernel<<<65536 / TM, 256, 0, stream>>>(
      s_u, s_i, s_user_table, s_item_table,
      esWT, es_b, ecWT, ec_b, adjT, stWT, st_b, sp_W, sp_b,
      cl1WT, cl1_b, cl2_W, cl2_b,
      o_s_pred, o_s_cls, o_s_inp, o_s_causal);
  cdcor_branch_kernel<<<65536 / TM, 256, 0, stream>>>(
      t_u, t_i, t_user_table, t_item_table,
      etWT, et_b, ecWT, ec_b, adjT, ttWT, tt_b, tp_W, tp_b,
      cl1WT, cl1_b, cl2_W, cl2_b,
      o_t_pred, o_t_cls, o_t_inp, o_t_causal);
}
